// Block_91173565760187
// MI455X (gfx1250) — matrix operand order repaired and checked by reading, not compiled, not run on hardware
//
#include <hip/hip_runtime.h>
#include <hip/hip_bf16.h>

// ---------------------------------------------------------------------------
// MI455X (gfx1250) transformer block: bf16 WMMA GEMMs + WMMA attention.
// ---------------------------------------------------------------------------

typedef __attribute__((ext_vector_type(16))) __bf16 v16bf;
typedef __attribute__((ext_vector_type(8)))  __bf16 v8bf;
typedef __attribute__((ext_vector_type(8)))  float  v8f;

#define NHEADS 12
#define DIMM   384
#define LTOK   8192        // T*FR = 128*64
#define MROWS  32768       // B*L

// ---- WMMA fragment loaders -------------------------------------------------
// A-matrix 16x32 bf16 layout (ISA 7.12.2): lane r (0..15) half h:
//   elements 0..7  -> K = k0 + h*8 + e
//   elements 8..15 -> K = k0 + 16 + h*8 + (e-8)
__device__ inline v16bf frag_load_a(const __bf16* base, int ld, int row0, int k0, int lane) {
  int h = lane >> 4, r = lane & 15;
  const __bf16* p = base + (size_t)(row0 + r) * ld + k0;
  union { v16bf v; v8bf q[2]; } u;
  u.q[0] = *(const v8bf*)(p + h * 8);
  u.q[1] = *(const v8bf*)(p + 16 + h * 8);
  return u.v;
}
// B-matrix 32x16 bf16 (stored as Bt: N x K row-major). Lane n holds column n:
//   half h: elements 0..7 -> K = k0 + h*8 + e, elements 8..15 -> K = k0 + 16 + h*8 + (e-8)
__device__ inline v16bf frag_load_b(const __bf16* base, int ld, int row0, int k0, int lane) {
  int h = lane >> 4, r = lane & 15;
  const __bf16* p = base + (size_t)(row0 + r) * ld + k0;
  union { v16bf v; v8bf q[2]; } u;
  u.q[0] = *(const v8bf*)(p + h * 8);
  u.q[1] = *(const v8bf*)(p + 16 + h * 8);
  return u.v;
}

// ---- generic bf16 GEMM: C = A(MxK, bf16) * Bt(NxK, bf16)^T -----------------
// epi 0: outF = acc + bias         (f32)
// epi 1: outF = resid + acc + bias (f32, resid may alias outF)
// epi 2: outB = gelu(acc + bias)   (bf16)
__global__ __launch_bounds__(256)
void gemm_kernel(const __bf16* __restrict__ A, const __bf16* __restrict__ Bt,
                 const float* __restrict__ bias, const float* __restrict__ resid,
                 float* __restrict__ outF, __bf16* __restrict__ outB,
                 int M, int N, int K, int epi)
{
  int tid = threadIdx.x, lane = tid & 31, w = tid >> 5;
  int m0 = blockIdx.x * 128 + w * 16;
  int n0 = blockIdx.y * 64;
  int h = lane >> 4, r = lane & 15;

  v8f acc[4] = {};
  for (int k0 = 0; k0 < K; k0 += 32) {
    // Batch ALL fragment loads into distinct registers first so the backend
    // can clause the global_load_b128s and use descending s_wait_loadcnt
    // instead of wait(0) before every wmma.
    v16bf af = frag_load_a(A, K, m0, k0, lane);
    v16bf bfr0 = frag_load_b(Bt, K, n0,      k0, lane);
    v16bf bfr1 = frag_load_b(Bt, K, n0 + 16, k0, lane);
    v16bf bfr2 = frag_load_b(Bt, K, n0 + 32, k0, lane);
    v16bf bfr3 = frag_load_b(Bt, K, n0 + 48, k0, lane);
    if (k0 + 32 < K)
      __builtin_prefetch(A + (size_t)(m0 + r) * K + k0 + 32, 0, 3);
    acc[0] = __builtin_amdgcn_wmma_f32_16x16x32_bf16(false, af, false, bfr0, (short)0, acc[0], false, false);
    acc[1] = __builtin_amdgcn_wmma_f32_16x16x32_bf16(false, af, false, bfr1, (short)0, acc[1], false, false);
    acc[2] = __builtin_amdgcn_wmma_f32_16x16x32_bf16(false, af, false, bfr2, (short)0, acc[2], false, false);
    acc[3] = __builtin_amdgcn_wmma_f32_16x16x32_bf16(false, af, false, bfr3, (short)0, acc[3], false, false);
  }
#pragma unroll
  for (int j = 0; j < 4; ++j) {
    int n = n0 + j * 16 + r;
    float bn = bias ? bias[n] : 0.0f;
#pragma unroll
    for (int e = 0; e < 8; ++e) {
      int m = m0 + e + 8 * h;
      size_t idx = (size_t)m * N + n;
      float val = acc[j][e] + bn;
      if (epi == 0) {
        outF[idx] = val;
      } else if (epi == 1) {
        outF[idx] = resid[idx] + val;
      } else {
        float t = 0.7978845608f * (val + 0.044715f * val * val * val);
        float g = 0.5f * val * (1.0f + tanhf(t));
        outB[idx] = (__bf16)g;
      }
    }
  }
}

// ---- weight transpose+cast: wt[n*K+k] = (bf16) w[k*N+n] --------------------
__global__ void wtrans_kernel(const float* __restrict__ w, __bf16* __restrict__ wt,
                              int K, int N)
{
  int i = blockIdx.x * blockDim.x + threadIdx.x;
  if (i >= K * N) return;
  int n = i / K, k = i - n * K;
  wt[i] = (__bf16)w[(size_t)k * N + n];
}

// ---- stage 1: (B,DIM,T,FR) -> xs (M,DIM) f32 + xn = rms(xs)*n1 bf16 --------
__global__ __launch_bounds__(384)
void pre_kernel(const float* __restrict__ x, const float* __restrict__ n1,
                float* __restrict__ xs, __bf16* __restrict__ xn)
{
  int token = blockIdx.x;            // 0..32767
  int b = token >> 13, l = token & (LTOK - 1);
  int c = threadIdx.x;               // 0..383
  float val = x[(size_t)b * DIMM * LTOK + (size_t)c * LTOK + l];
  __shared__ float red[DIMM];
  red[c] = val * val;
  __syncthreads();
  if (c < 128) red[c] += red[c + 128] + red[c + 256];
  __syncthreads();
  for (int st = 64; st > 0; st >>= 1) {
    if (c < st) red[c] += red[c + st];
    __syncthreads();
  }
  float scale = rsqrtf(red[0] * (1.0f / DIMM) + 1e-6f);
  xs[(size_t)token * DIMM + c] = val;
  xn[(size_t)token * DIMM + c] = (__bf16)(val * scale * n1[c]);
}

// ---- stage 3: qkv f32 -> rms(q),rms(k),rope(q),rope(k) -> bf16 q,k,v -------
__global__ __launch_bounds__(384)
void qkv_post_kernel(const float* __restrict__ qkv,
                     const float* __restrict__ nq, const float* __restrict__ nk,
                     const float* __restrict__ cosT, const float* __restrict__ sinT,
                     __bf16* __restrict__ qo, __bf16* __restrict__ ko,
                     __bf16* __restrict__ vo)
{
  int token = blockIdx.x;
  int l = token & (LTOK - 1);
  int c = threadIdx.x;
  size_t base = (size_t)token * 1152;
  float qv = qkv[base + c];
  float kv = qkv[base + 384 + c];
  float vv = qkv[base + 768 + c];

  __shared__ float red[DIMM];
  __shared__ float shq[DIMM];
  __shared__ float shk[DIMM];

  // rms(q)
  red[c] = qv * qv;
  __syncthreads();
  if (c < 128) red[c] += red[c + 128] + red[c + 256];
  __syncthreads();
  for (int st = 64; st > 0; st >>= 1) { if (c < st) red[c] += red[c + st]; __syncthreads(); }
  float qs = rsqrtf(red[0] * (1.0f / DIMM) + 1e-6f);
  __syncthreads();
  // rms(k)
  red[c] = kv * kv;
  __syncthreads();
  if (c < 128) red[c] += red[c + 128] + red[c + 256];
  __syncthreads();
  for (int st = 64; st > 0; st >>= 1) { if (c < st) red[c] += red[c + st]; __syncthreads(); }
  float ks = rsqrtf(red[0] * (1.0f / DIMM) + 1e-6f);

  shq[c] = qv * qs * nq[c];
  shk[c] = kv * ks * nk[c];
  __syncthreads();

  int i = (c & 31) >> 1;   // rotary frequency index inside the head
  float cv = cosT[(size_t)l * 16 + i];
  float sv = sinT[(size_t)l * 16 + i];
  float qr, kr;
  if ((c & 1) == 0) {
    qr = shq[c] * cv - shq[c + 1] * sv;
    kr = shk[c] * cv - shk[c + 1] * sv;
  } else {
    qr = shq[c - 1] * sv + shq[c] * cv;
    kr = shk[c - 1] * sv + shk[c] * cv;
  }
  size_t o = (size_t)token * DIMM + c;
  qo[o] = (__bf16)qr;
  ko[o] = (__bf16)kr;
  vo[o] = (__bf16)vv;
}

// ---- attention token-index maps for the 3 reshape patterns -----------------
template<int PAT>
__device__ inline size_t map_token(int g, int s) {
  if (PAT == 0) {        // to_t: g = b*128 + t, s = f (0..63)
    int b = g >> 7, t = g & 127;
    return (size_t)b * LTOK + t * 64 + s;
  } else if (PAT == 1) { // to_f: g = b*64 + f, s = t (0..127)
    int b = g >> 6, f = g & 63;
    return (size_t)b * LTOK + s * 64 + f;
  } else {               // to_blk: g = ((b*32)+t1)*8 + f1, s = t2*8+f2 (0..31)
    int b = g >> 8, t1 = (g >> 3) & 31, f1 = g & 7;
    int t2 = s >> 3, f2 = s & 7;
    return (size_t)b * LTOK + (t1 * 4 + t2) * 64 + f1 * 8 + f2;
  }
}

// ---- attention: one block per (group g, head). head_dim=32 = one WMMA K-step
template<int PAT, int S>
__global__ __launch_bounds__(128)
void attn_kernel(const __bf16* __restrict__ Q, const __bf16* __restrict__ Kb,
                 const __bf16* __restrict__ Vb, float* __restrict__ Y)
{
  int head = blockIdx.x % NHEADS;
  int g = blockIdx.x / NHEADS;
  int tid = threadIdx.x, lane = tid & 31, w = tid >> 5;
  int h = lane >> 4, r = lane & 15;

  __shared__ __align__(16) __bf16 Vt[32 * 128];     // V^T: [d][s]
  __shared__ __align__(16) float  Sb[4][16 * 128];  // per-wave score rows
  __shared__ __align__(16) __bf16 Pb[4][16 * 128];  // per-wave softmax rows

  for (int i = tid; i < 32 * S; i += 128) {
    int d = i & 31, s = i >> 5;
    Vt[d * 128 + s] = Vb[map_token<PAT>(g, s) * DIMM + head * 32 + d];
  }
  __syncthreads();

  const float scale = 0.17677669529663687f;   // 1/sqrt(32)
  for (int qb = w * 16; qb < S; qb += 64) {
    // ---- S = Q K^T : one wmma per 16x16 tile (K=head_dim=32) ----
    size_t qtok = map_token<PAT>(g, qb + r);
    const __bf16* qp = Q + qtok * DIMM + head * 32;
    union { v16bf v; v8bf q[2]; } uq;
    uq.q[0] = *(const v8bf*)(qp + h * 8);
    uq.q[1] = *(const v8bf*)(qp + 16 + h * 8);
    v16bf qf = uq.v;

    float* Srow = &Sb[w][0];
#pragma unroll
    for (int jt = 0; jt < S / 16; ++jt) {
      size_t ktok = map_token<PAT>(g, jt * 16 + r);
      const __bf16* kp = Kb + ktok * DIMM + head * 32;
      union { v16bf v; v8bf q[2]; } uk;
      uk.q[0] = *(const v8bf*)(kp + h * 8);
      uk.q[1] = *(const v8bf*)(kp + 16 + h * 8);
      v8f sacc = {};
      sacc = __builtin_amdgcn_wmma_f32_16x16x32_bf16(
          false, qf, false, uk.v, (short)0, sacc, false, false);
#pragma unroll
      for (int e = 0; e < 8; ++e)
        Srow[(e + 8 * h) * 128 + jt * 16 + r] = sacc[e] * scale;
    }

    // ---- softmax: lanes 0..15 each own one of this wave's 16 rows ----
    if (lane < 16) {
      float* row = Srow + lane * 128;
      float mx = -3.0e38f;
      for (int s2 = 0; s2 < S; ++s2) mx = fmaxf(mx, row[s2]);
      float sum = 0.0f;
      for (int s2 = 0; s2 < S; ++s2) { float e = __expf(row[s2] - mx); row[s2] = e; sum += e; }
      float inv = 1.0f / sum;
      __bf16* prow = &Pb[w][lane * 128];
      for (int s2 = 0; s2 < S; ++s2) prow[s2] = (__bf16)(row[s2] * inv);
    }

    // ---- O = P V : N=32 (2 tiles), K=S ----
    v8f o0 = {}, o1 = {};
#pragma unroll
    for (int kt = 0; kt < S; kt += 32) {
      v16bf pf  = frag_load_a(&Pb[w][0], 128, 0, kt, lane);
      v16bf vf0 = frag_load_b(Vt, 128, 0, kt, lane);
      v16bf vf1 = frag_load_b(Vt, 128, 16, kt, lane);
      o0 = __builtin_amdgcn_wmma_f32_16x16x32_bf16(false, pf, false, vf0, (short)0, o0, false, false);
      o1 = __builtin_amdgcn_wmma_f32_16x16x32_bf16(false, pf, false, vf1, (short)0, o1, false, false);
    }
#pragma unroll
    for (int e = 0; e < 8; ++e) {
      size_t ot = map_token<PAT>(g, qb + e + 8 * h);
      float* yp = Y + ot * DIMM + head * 32;
      yp[r] = o0[e];
      yp[16 + r] = o1[e];
    }
  }
}

// ---- y1+y2+y3 -> bf16 ------------------------------------------------------
__global__ void combine_kernel(const float* __restrict__ y1, const float* __restrict__ y2,
                               const float* __restrict__ y3, __bf16* __restrict__ out, int n)
{
  int i = blockIdx.x * blockDim.x + threadIdx.x;
  if (i < n) out[i] = (__bf16)(y1[i] + y2[i] + y3[i]);
}

// ---- rms(xs)*n2 -> bf16 ----------------------------------------------------
__global__ __launch_bounds__(384)
void rms2_kernel(const float* __restrict__ xs, const float* __restrict__ n2,
                 __bf16* __restrict__ hn)
{
  int token = blockIdx.x;
  int c = threadIdx.x;
  float val = xs[(size_t)token * DIMM + c];
  __shared__ float red[DIMM];
  red[c] = val * val;
  __syncthreads();
  if (c < 128) red[c] += red[c + 128] + red[c + 256];
  __syncthreads();
  for (int st = 64; st > 0; st >>= 1) { if (c < st) red[c] += red[c + st]; __syncthreads(); }
  float scale = rsqrtf(red[0] * (1.0f / DIMM) + 1e-6f);
  hn[(size_t)token * DIMM + c] = (__bf16)(val * scale * n2[c]);
}

// ---- xs (M,DIM) -> out (B,DIM,T,FR) ----------------------------------------
__global__ void final_kernel(const float* __restrict__ xs, float* __restrict__ out, int n)
{
  int i = blockIdx.x * blockDim.x + threadIdx.x;
  if (i >= n) return;
  int b = i / (DIMM * LTOK);
  int rem = i - b * (DIMM * LTOK);
  int c = rem / LTOK;
  int l = rem - c * LTOK;
  out[i] = xs[((size_t)b * LTOK + l) * DIMM + c];
}

// ---------------------------------------------------------------------------
extern "C" void kernel_launch(void* const* d_in, const int* in_sizes, int n_in,
                              void* d_out, int out_size, void* d_ws, size_t ws_size,
                              hipStream_t stream) {
  const float* x      = (const float*)d_in[0];
  const float* cosT   = (const float*)d_in[1];
  const float* sinT   = (const float*)d_in[2];
  const float* qkv_w  = (const float*)d_in[3];
  const float* qkv_b  = (const float*)d_in[4];
  const float* nq_s   = (const float*)d_in[5];
  const float* nk_s   = (const float*)d_in[6];
  const float* proj_w = (const float*)d_in[7];
  const float* proj_b = (const float*)d_in[8];
  const float* n1_s   = (const float*)d_in[9];
  const float* n2_s   = (const float*)d_in[10];
  const float* w1     = (const float*)d_in[11];
  const float* b1     = (const float*)d_in[12];
  const float* w2     = (const float*)d_in[13];
  const float* b2     = (const float*)d_in[14];

  char* ws = (char*)d_ws;
  const size_t SZ_XS  = (size_t)MROWS * DIMM * 4;   // 50331648
  const size_t SZ_ACT = (size_t)MROWS * DIMM * 2;   // 25165824
  const size_t SZ_BIG = (size_t)MROWS * 1152 * 4;   // 150994944 (= 3 * SZ_XS)
  size_t o = 0;
  float*  xs    = (float*)(ws + o);  o += SZ_XS;
  __bf16* act   = (__bf16*)(ws + o); o += SZ_ACT;   // xn -> ysum -> hn (sequential reuse)
  char*   big   = ws + o;            o += SZ_BIG;   // qkv f32 -> y1/y2/y3 -> hact
  __bf16* qb    = (__bf16*)(ws + o); o += SZ_ACT;
  __bf16* kb    = (__bf16*)(ws + o); o += SZ_ACT;
  __bf16* vb    = (__bf16*)(ws + o); o += SZ_ACT;
  __bf16* wqkvT = (__bf16*)(ws + o); o += (size_t)1152 * 384 * 2;
  __bf16* wprojT= (__bf16*)(ws + o); o += (size_t)384 * 384 * 2;
  __bf16* w1T   = (__bf16*)(ws + o); o += (size_t)1536 * 384 * 2;
  __bf16* w2T   = (__bf16*)(ws + o); o += (size_t)384 * 1536 * 2;

  float*  qkvf = (float*)big;
  float*  y1   = (float*)big;
  float*  y2   = y1 + (size_t)MROWS * DIMM;
  float*  y3   = y2 + (size_t)MROWS * DIMM;
  __bf16* hact = (__bf16*)big;

  // 0) weights -> bf16, transposed to (N, K)
  wtrans_kernel<<<(384 * 1152 + 255) / 256, 256, 0, stream>>>(qkv_w, wqkvT, 384, 1152);
  wtrans_kernel<<<(384 * 384 + 255) / 256, 256, 0, stream>>>(proj_w, wprojT, 384, 384);
  wtrans_kernel<<<(384 * 1536 + 255) / 256, 256, 0, stream>>>(w1, w1T, 384, 1536);
  wtrans_kernel<<<(1536 * 384 + 255) / 256, 256, 0, stream>>>(w2, w2T, 1536, 384);

  // 1) transpose input + RMSNorm(n1) -> xs f32, act bf16
  pre_kernel<<<MROWS, 384, 0, stream>>>(x, n1_s, xs, act);

  // 2) QKV GEMM (M=32768, N=1152, K=384), +bias, f32 out
  gemm_kernel<<<dim3(MROWS / 128, 1152 / 64), 256, 0, stream>>>(
      act, wqkvT, qkv_b, nullptr, qkvf, nullptr, MROWS, 1152, 384, 0);

  // 3) rms(q/k) + rope -> bf16 q,k,v
  qkv_post_kernel<<<MROWS, 384, 0, stream>>>(qkvf, nq_s, nk_s, cosT, sinT, qb, kb, vb);

  // 4) three attention patterns (y* reuse the qkv region)
  attn_kernel<0, 64 ><<<512  * NHEADS, 128, 0, stream>>>(qb, kb, vb, y1);
  attn_kernel<1, 128><<<256  * NHEADS, 128, 0, stream>>>(qb, kb, vb, y2);
  attn_kernel<2, 32 ><<<1024 * NHEADS, 128, 0, stream>>>(qb, kb, vb, y3);

  // 5) y1+y2+y3 -> bf16 (into act)
  combine_kernel<<<(MROWS * DIMM + 255) / 256, 256, 0, stream>>>(
      y1, y2, y3, act, MROWS * DIMM);

  // 6) proj GEMM + bias + residual (in-place on xs)
  gemm_kernel<<<dim3(MROWS / 128, 384 / 64), 256, 0, stream>>>(
      act, wprojT, proj_b, xs, xs, nullptr, MROWS, 384, 384, 1);

  // 7) RMSNorm(n2) -> act bf16
  rms2_kernel<<<MROWS, 384, 0, stream>>>(xs, n2_s, act);

  // 8) MLP up GEMM + bias + gelu -> hact bf16 (reuses big region)
  gemm_kernel<<<dim3(MROWS / 128, 1536 / 64), 256, 0, stream>>>(
      act, w1T, b1, nullptr, nullptr, hact, MROWS, 1536, 384, 2);

  // 9) MLP down GEMM + bias + residual (in-place on xs)
  gemm_kernel<<<dim3(MROWS / 128, 384 / 64), 256, 0, stream>>>(
      hact, w2T, b2, xs, xs, nullptr, MROWS, 384, 1536, 1);

  // 10) xs -> (B, DIM, T, FR)
  final_kernel<<<(MROWS * DIMM + 255) / 256, 256, 0, stream>>>(
      xs, (float*)d_out, MROWS * DIMM);
}
